// FieldNet_9216999817554
// MI455X (gfx1250) — matrix operand order repaired and checked by reading, not compiled, not run on hardware
//
#include <hip/hip_runtime.h>
#include <hip/hip_bf16.h>
#include <math.h>

typedef __attribute__((ext_vector_type(16))) __bf16 v16bf;
typedef __attribute__((ext_vector_type(8)))  float  v8f;

#define NN 20000
#define EE 320000
#define NBINS 32
#define LAYERS 3
#define CUTR 5.0f

// B chunks staged in LDS with padded stride (40 bf16 = 20 dwords) to spread banks
#define SB_STRIDE 40
// hidden tile stride (136 bf16 = 68 dwords -> row stride 4 banks)
#define HID_STRIDE 136

#if defined(__AMDGCN__) && __has_builtin(__builtin_amdgcn_global_load_async_to_lds_b128)
#define ASYNC_OK 1
typedef int v4i_vs __attribute__((vector_size(16)));
typedef __attribute__((address_space(1))) v4i_vs* as1_v4i;
typedef __attribute__((address_space(3))) v4i_vs* as3_v4i;
#else
#define ASYNC_OK 0
#endif

__device__ __forceinline__ float swishf(float x) {
  return x / (1.0f + __expf(-x));
}

__device__ __forceinline__ void wait_async0() {
#if ASYNC_OK
#if __has_builtin(__builtin_amdgcn_s_wait_asynccnt)
  __builtin_amdgcn_s_wait_asynccnt(0);
#else
  asm volatile("s_wait_asynccnt 0x0" ::: "memory");
#endif
#endif
}

// Stage one 32xK-chunk of the (pre-transposed, bf16, [col][K]) weight matrix
// into LDS layout sB[col*SB_STRIDE + k]. 512 x 16B copies over 256 threads.
__device__ __forceinline__ void stage_B(__bf16* sB, const __bf16* __restrict__ Bt,
                                        int K, int k0, int tid) {
  for (int i = tid; i < 512; i += 256) {
    int col = i >> 2, q = i & 3;
    const __bf16* g = Bt + (size_t)col * K + k0 + q * 8;
    __bf16* l = sB + col * SB_STRIDE + q * 8;
#if ASYNC_OK
    __builtin_amdgcn_global_load_async_to_lds_b128((as1_v4i)(void*)g, (as3_v4i)(void*)l, 0, 0);
#else
    *reinterpret_cast<uint4*>(l) = *reinterpret_cast<const uint4*>(g);
#endif
  }
  wait_async0();
}

// Load the 16x32 bf16 A-fragment (from fp32 global) for this lane.
__device__ __forceinline__ v16bf load_afrag_f32(const float* __restrict__ ap, bool inr) {
  v16bf a;
  #pragma unroll
  for (int e = 0; e < 8; ++e) a[e]     = inr ? (__bf16)ap[e]      : (__bf16)0.0f;
  #pragma unroll
  for (int e = 0; e < 8; ++e) a[e + 8] = inr ? (__bf16)ap[16 + e] : (__bf16)0.0f;
  return a;
}

__device__ __forceinline__ v16bf load_afrag_lds(const __bf16* ap) {
  v16bf a;
  #pragma unroll
  for (int e = 0; e < 8; ++e) a[e]     = ap[e];
  #pragma unroll
  for (int e = 0; e < 8; ++e) a[e + 8] = ap[16 + e];
  return a;
}

__device__ __forceinline__ v16bf load_bfrag(const __bf16* sB, int ct, int rloc, int hsel) {
  const __bf16* bp = sB + (ct * 16 + rloc) * SB_STRIDE + hsel * 8;
  v16bf b;
  #pragma unroll
  for (int e = 0; e < 8; ++e) b[e] = bp[e];
  #pragma unroll
  for (int e = 0; e < 8; ++e) b[e + 8] = bp[16 + e];
  return b;
}

// ---------------------------------------------------------------------------
// Single GEMM:  C[M,128] = (act? swish : id)(A[M,K] @ B + bias) * rowscale
// B passed pre-transposed bf16 [col][K].
// ---------------------------------------------------------------------------
__global__ __launch_bounds__(256)
void k_gemm128(const float* __restrict__ A, const __bf16* __restrict__ Bt,
               const float* __restrict__ bias, const float* __restrict__ rowscale,
               float* __restrict__ C, int M, int K, int act)
{
  __shared__ __attribute__((aligned(16))) __bf16 sB[128 * SB_STRIDE];
  const int tid = threadIdx.x;
  const int wave = tid >> 5, lane = tid & 31;
  const int rloc = lane & 15, hsel = lane >> 4;
  const int rowBase = blockIdx.x * 128 + wave * 16;
  const int rowA = rowBase + rloc;
  const bool inr = rowA < M;

  v8f acc[8];
  #pragma unroll
  for (int ct = 0; ct < 8; ++ct) {
    #pragma unroll
    for (int i = 0; i < 8; ++i) acc[ct][i] = 0.0f;
  }

  for (int k0 = 0; k0 < K; k0 += 32) {
    stage_B(sB, Bt, K, k0, tid);
    __syncthreads();
    const float* ap = A + (size_t)rowA * K + k0 + hsel * 8;
    if (k0 + 32 < K) __builtin_prefetch(ap + 32, 0, 0);
    v16bf afrag = load_afrag_f32(ap, inr);
    #pragma unroll
    for (int ct = 0; ct < 8; ++ct) {
      v16bf bfrag = load_bfrag(sB, ct, rloc, hsel);
      acc[ct] = __builtin_amdgcn_wmma_f32_16x16x32_bf16(
          false, afrag, false, bfrag, (short)0, acc[ct], false, false);
    }
    __syncthreads();
  }

  #pragma unroll
  for (int ct = 0; ct < 8; ++ct) {
    int col = ct * 16 + rloc;
    float bi = bias ? bias[col] : 0.0f;
    #pragma unroll
    for (int r = 0; r < 8; ++r) {
      int row = rowBase + r + hsel * 8;
      if (row < M) {
        float v = acc[ct][r] + bi;
        if (act) v = swishf(v);
        if (rowscale) v *= rowscale[row];
        C[(size_t)row * 128 + col] = v;
      }
    }
  }
}

// ---------------------------------------------------------------------------
// Fused 2-layer MLP:  C[M,128] = swish(A[M,K1] @ B1 + b1) @ B2 + b2
// Hidden activation kept in LDS as bf16 (never touches HBM).
// ---------------------------------------------------------------------------
__global__ __launch_bounds__(256)
void k_mlp2(const float* __restrict__ A, const __bf16* __restrict__ B1t,
            const float* __restrict__ b1, const __bf16* __restrict__ B2t,
            const float* __restrict__ b2, float* __restrict__ C, int M, int K1)
{
  __shared__ __attribute__((aligned(16))) __bf16 sB[128 * SB_STRIDE];
  __shared__ __attribute__((aligned(16))) __bf16 hid[128 * HID_STRIDE];
  const int tid = threadIdx.x;
  const int wave = tid >> 5, lane = tid & 31;
  const int rloc = lane & 15, hsel = lane >> 4;
  const int rowBase = blockIdx.x * 128 + wave * 16;
  const int rowA = rowBase + rloc;
  const bool inr = rowA < M;

  v8f acc[8];
  #pragma unroll
  for (int ct = 0; ct < 8; ++ct) {
    #pragma unroll
    for (int i = 0; i < 8; ++i) acc[ct][i] = 0.0f;
  }

  // ---- phase 1: hidden = swish(A @ B1 + b1) -> LDS (bf16) ----
  for (int k0 = 0; k0 < K1; k0 += 32) {
    stage_B(sB, B1t, K1, k0, tid);
    __syncthreads();
    const float* ap = A + (size_t)rowA * K1 + k0 + hsel * 8;
    if (k0 + 32 < K1) __builtin_prefetch(ap + 32, 0, 0);
    v16bf afrag = load_afrag_f32(ap, inr);
    #pragma unroll
    for (int ct = 0; ct < 8; ++ct) {
      v16bf bfrag = load_bfrag(sB, ct, rloc, hsel);
      acc[ct] = __builtin_amdgcn_wmma_f32_16x16x32_bf16(
          false, afrag, false, bfrag, (short)0, acc[ct], false, false);
    }
    __syncthreads();
  }
  #pragma unroll
  for (int ct = 0; ct < 8; ++ct) {
    int col = ct * 16 + rloc;
    float bi = b1[col];
    #pragma unroll
    for (int r = 0; r < 8; ++r) {
      int lrow = wave * 16 + r + hsel * 8;
      hid[lrow * HID_STRIDE + col] = (__bf16)swishf(acc[ct][r] + bi);
    }
  }
  __syncthreads();

  // ---- phase 2: C = hidden @ B2 + b2 ----
  #pragma unroll
  for (int ct = 0; ct < 8; ++ct) {
    #pragma unroll
    for (int i = 0; i < 8; ++i) acc[ct][i] = 0.0f;
  }
  const __bf16* hrow = &hid[(wave * 16 + rloc) * HID_STRIDE];
  for (int k0 = 0; k0 < 128; k0 += 32) {
    stage_B(sB, B2t, 128, k0, tid);
    __syncthreads();
    v16bf afrag = load_afrag_lds(hrow + k0 + hsel * 8);
    #pragma unroll
    for (int ct = 0; ct < 8; ++ct) {
      v16bf bfrag = load_bfrag(sB, ct, rloc, hsel);
      acc[ct] = __builtin_amdgcn_wmma_f32_16x16x32_bf16(
          false, afrag, false, bfrag, (short)0, acc[ct], false, false);
    }
    __syncthreads();
  }
  #pragma unroll
  for (int ct = 0; ct < 8; ++ct) {
    int col = ct * 16 + rloc;
    float bi = b2 ? b2[col] : 0.0f;
    #pragma unroll
    for (int r = 0; r < 8; ++r) {
      int row = rowBase + r + hsel * 8;
      if (row < M) C[(size_t)row * 128 + col] = acc[ct][r] + bi;
    }
  }
}

// ---------------------------------------------------------------------------
// Weight conversion: fp32 [m][K][128] -> bf16 transposed [m][col][K]
// ---------------------------------------------------------------------------
__global__ void k_wcvt(const float* __restrict__ B, __bf16* __restrict__ Bt,
                       int K, int total) {
  int i = blockIdx.x * blockDim.x + threadIdx.x;
  if (i >= total) return;
  int per = 128 * K;
  int m = i / per, r = i % per;
  int col = r / K, k = r % K;
  Bt[i] = (__bf16)B[(size_t)m * per + (size_t)k * 128 + col];
}

// ---------------------------------------------------------------------------
// Elementwise / geometry / scatter kernels (f32 precision paths)
// ---------------------------------------------------------------------------
__global__ void k_zero(float* __restrict__ p, size_t n) {
  size_t i = (size_t)blockIdx.x * blockDim.x + threadIdx.x;
  if (i < n) p[i] = 0.0f;
}

__global__ void k_gather(const int* __restrict__ z, const float* __restrict__ emb,
                         float* __restrict__ feat, int n) {
  int i = blockIdx.x * blockDim.x + threadIdx.x;
  if (i >= n * 128) return;
  int node = i >> 7, d = i & 127;
  feat[i] = emb[z[node] * 128 + d];
}

__global__ void k_edge_geom(const float* __restrict__ pos, const int* __restrict__ src,
                            const int* __restrict__ dst, float* __restrict__ rij_out,
                            float* __restrict__ Cr, float* __restrict__ vhat,
                            float* __restrict__ fij, int E) {
  int e = blockIdx.x * blockDim.x + threadIdx.x;
  if (e >= E) return;
  int s = src[e], d = dst[e];
  float vx = pos[d * 3 + 0] - pos[s * 3 + 0];
  float vy = pos[d * 3 + 1] - pos[s * 3 + 1];
  float vz = pos[d * 3 + 2] - pos[s * 3 + 2];
  float r = sqrtf(vx * vx + vy * vy + vz * vz + 1e-12f);
  rij_out[e] = r;
  float t = r / CUTR;
  float r2 = t * t;
  float c = 0.0f;
  if (r2 < 1.0f) c = __expf(1.0f - 1.0f / (1.0f - r2));   // mollifier cutoff
  Cr[e] = c;
  float inv1r = 1.0f / (1.0f + r);
  vhat[e * 3 + 0] = vx * inv1r;
  vhat[e * 3 + 1] = vy * inv1r;
  vhat[e * 3 + 2] = vz * inv1r;
  const float dlt = CUTR / (float)(NBINS - 1);
  const float gco = 0.5f / (dlt * dlt);
  #pragma unroll
  for (int j = 0; j < NBINS; ++j) {
    float df = r - (float)j * dlt;
    fij[(size_t)e * NBINS + j] = __expf(-gco * df * df);
  }
}

__global__ void k_scatter_feat(const float* __restrict__ x, const float* __restrict__ W,
                               const float* __restrict__ Cr, const int* __restrict__ src,
                               const int* __restrict__ dst, float* __restrict__ agg) {
  int e = blockIdx.x, f = threadIdx.x;
  int s = src[e], d = dst[e];
  float v = x[(size_t)s * 128 + f] * W[(size_t)e * 128 + f] * Cr[e];
  atomicAdd(&agg[(size_t)d * 128 + f], v);
}

__global__ void k_scatter_dipole(const float* __restrict__ h, const float* __restrict__ Cr,
                                 const float* __restrict__ vhat, const int* __restrict__ src,
                                 const int* __restrict__ dst, float* __restrict__ mu) {
  int e = blockIdx.x, f = threadIdx.x;
  int s = src[e], d = dst[e];
  float m = h[(size_t)s * 128 + f] * Cr[e];
  float v0 = vhat[e * 3 + 0], v1 = vhat[e * 3 + 1], v2 = vhat[e * 3 + 2];
  size_t base = ((size_t)d * 128 + f) * 3;
  atomicAdd(&mu[base + 0], m * v0);
  atomicAdd(&mu[base + 1], m * v1);
  atomicAdd(&mu[base + 2], m * v2);
}

__global__ void k_scatter_tdot(const float* __restrict__ mu, const float* __restrict__ vhat,
                               const float* __restrict__ Wt, const int* __restrict__ src,
                               const int* __restrict__ dst, float* __restrict__ s_agg) {
  int e = blockIdx.x, f = threadIdx.x;
  int s = src[e], d = dst[e];
  size_t base = ((size_t)s * 128 + f) * 3;
  float dot = mu[base + 0] * vhat[e * 3 + 0] +
              mu[base + 1] * vhat[e * 3 + 1] +
              mu[base + 2] * vhat[e * 3 + 2];
  atomicAdd(&s_agg[(size_t)d * 128 + f], dot * Wt[(size_t)e * 128 + f]);
}

__global__ void k_build_cat(const float* __restrict__ feat, const float* __restrict__ mu,
                            float* __restrict__ cat, int n) {
  int i = blockIdx.x * blockDim.x + threadIdx.x;
  if (i >= n * 128) return;
  int node = i >> 7, f = i & 127;
  cat[(size_t)node * 256 + f] = feat[i];
  size_t b = (size_t)i * 3;
  float m0 = mu[b], m1 = mu[b + 1], m2 = mu[b + 2];
  cat[(size_t)node * 256 + 128 + f] = m0 * m0 + m1 * m1 + m2 * m2;
}

__global__ void k_combine(float* __restrict__ vtotal, const float* __restrict__ mmuls,
                          const float* __restrict__ vdip, float* __restrict__ feat, int n) {
  int i = blockIdx.x * blockDim.x + threadIdx.x;
  if (i >= n * 128) return;
  float vt = vtotal[i] + mmuls[i] + vdip[i];
  vtotal[i] = vt;
  feat[i] += vt;
}

// ---------------------------------------------------------------------------
extern "C" void kernel_launch(void* const* d_in, const int* in_sizes, int n_in,
                              void* d_out, int out_size, void* d_ws, size_t ws_size,
                              hipStream_t stream) {
  const int N = NN, E = EE;
  const int*   z    = (const int*)  d_in[0];
  const float* pos  = (const float*)d_in[1];
  const int*   src  = (const int*)  d_in[2];
  const int*   dst  = (const int*)  d_in[3];
  const float* emb  = (const float*)d_in[4];
  const float* Wd   = (const float*)d_in[5];
  const float* bd   = (const float*)d_in[6];
  const float* Win  = (const float*)d_in[7];
  const float* fW1  = (const float*)d_in[8];
  const float* fb1  = (const float*)d_in[9];
  const float* fW2  = (const float*)d_in[10];
  const float* fb2  = (const float*)d_in[11];
  const float* oW1  = (const float*)d_in[12];
  const float* ob1  = (const float*)d_in[13];
  const float* oW2  = (const float*)d_in[14];
  const float* ob2  = (const float*)d_in[15];
  const float* mW1  = (const float*)d_in[16];
  const float* mb1  = (const float*)d_in[17];
  const float* mWs  = (const float*)d_in[18];
  const float* tfW1 = (const float*)d_in[19];
  const float* tfb1 = (const float*)d_in[20];
  const float* tWs  = (const float*)d_in[21];
  const float* tbs  = (const float*)d_in[22];

  // outputs: feat [N,128] | mu [N,128,3] | rij [E]
  float* feat = (float*)d_out;
  float* mu   = feat + (size_t)N * 128;
  float* rij  = mu + (size_t)N * 128 * 3;

  // ---- workspace carving ----
  float* w = (float*)d_ws;
  size_t off = 0;
  auto wsalloc = [&](size_t n) { float* p = w + off; off += n; return p; };
  float* Cr   = wsalloc((size_t)E);
  float* vhat = wsalloc((size_t)E * 3);
  float* fij  = wsalloc((size_t)E * NBINS);
  float* eb1  = wsalloc((size_t)E * 128);   // Wt
  float* eb2  = wsalloc((size_t)E * 128);   // W
  float* xb   = wsalloc((size_t)N * 128);
  float* agg  = wsalloc((size_t)N * 128);
  float* hn   = wsalloc((size_t)N * 128);
  float* vt   = wsalloc((size_t)N * 128);
  float* mm   = wsalloc((size_t)N * 128);
  float* sg   = wsalloc((size_t)N * 128);
  float* vd   = wsalloc((size_t)N * 128);
  float* cat  = wsalloc((size_t)N * 256);

  // bf16 transposed weight region
  __bf16* bw = (__bf16*)(w + off);
  size_t boff = 0;
  auto balloc = [&](size_t n) { __bf16* p = bw + boff; boff += n; return p; };
  __bf16* Wdt   = balloc((size_t)4 * 16384);
  __bf16* Wint  = balloc((size_t)3 * 16384);
  __bf16* fW1t  = balloc((size_t)3 * 4096);
  __bf16* fW2t  = balloc((size_t)3 * 16384);
  __bf16* oW1t  = balloc((size_t)3 * 16384);
  __bf16* oW2t  = balloc((size_t)3 * 16384);
  __bf16* mW1t  = balloc((size_t)3 * 32768);
  __bf16* mWst  = balloc((size_t)3 * 16384);
  __bf16* tfW1t = balloc((size_t)3 * 4096);
  __bf16* tWst  = balloc((size_t)3 * 16384);
  (void)ws_size; (void)n_in; (void)in_sizes; (void)out_size;

  auto wcvt = [&](const float* s, __bf16* dst2, int K, int nmat) {
    int total = nmat * 128 * K;
    k_wcvt<<<dim3((total + 255) / 256), dim3(256), 0, stream>>>(s, dst2, K, total);
  };
  auto gemm = [&](const float* A, const __bf16* Bt, const float* bias, const float* rs,
                  float* C, int M, int K, int act) {
    k_gemm128<<<dim3((M + 127) / 128), dim3(256), 0, stream>>>(A, Bt, bias, rs, C, M, K, act);
  };
  auto mlp2 = [&](const float* A, const __bf16* B1t, const float* b1,
                  const __bf16* B2t, const float* b2, float* C, int M, int K1) {
    k_mlp2<<<dim3((M + 127) / 128), dim3(256), 0, stream>>>(A, B1t, b1, B2t, b2, C, M, K1);
  };
  auto zero = [&](float* p, size_t n) {
    k_zero<<<dim3((unsigned)((n + 255) / 256)), dim3(256), 0, stream>>>(p, n);
  };

  // ---- weight preconversion (cheap: ~1.9 MB fp32 in, ~1 MB bf16 out) ----
  wcvt(Wd,   Wdt,   128, 4);
  wcvt(Win,  Wint,  128, 3);
  wcvt(fW1,  fW1t,  32,  3);
  wcvt(fW2,  fW2t,  128, 3);
  wcvt(oW1,  oW1t,  128, 3);
  wcvt(oW2,  oW2t,  128, 3);
  wcvt(mW1,  mW1t,  256, 3);
  wcvt(mWs,  mWst,  128, 3);
  wcvt(tfW1, tfW1t, 32,  3);
  wcvt(tWs,  tWst,  128, 3);

  // ---- setup: embeddings, geometry, mu = dipole(0, feat) ----
  zero(mu, (size_t)N * 128 * 3);
  k_gather<<<dim3((N * 128 + 255) / 256), dim3(256), 0, stream>>>(z, emb, feat, N);
  k_edge_geom<<<dim3((E + 127) / 128), dim3(128), 0, stream>>>(pos, src, dst, rij, Cr, vhat, fij, E);

  gemm(feat, Wdt, bd, nullptr, hn, N, 128, 1);                      // swish(feat @ Wd0 + bd0)
  k_scatter_dipole<<<dim3(E), dim3(128), 0, stream>>>(hn, Cr, vhat, src, dst, mu);

  for (int l = 0; l < LAYERS; ++l) {
    // cfconv filter net (fused 2-layer MLP, hidden stays in LDS)
    mlp2(fij, fW1t + (size_t)l * 4096, fb1 + l * 128,
         fW2t + (size_t)l * 16384, fb2 + l * 128, eb2, E, 32);
    // x = feat @ W_in ; agg = seg(x[src]*W*Cr)
    gemm(feat, Wint + (size_t)l * 16384, nullptr, nullptr, xb, N, 128, 0);
    zero(agg, (size_t)N * 128);
    k_scatter_feat<<<dim3(E), dim3(128), 0, stream>>>(xb, eb2, Cr, src, dst, agg);
    // v_total = swish(agg@oW1+ob1)@oW2 + ob2   (fused)
    mlp2(agg, oW1t + (size_t)l * 16384, ob1 + l * 128,
         oW2t + (size_t)l * 16384, ob2 + l * 128, vt, N, 128);
    // multipole interaction: mm = swish([feat,|mu|^2]@mW1+mb1)@mWs   (fused)
    k_build_cat<<<dim3((N * 128 + 255) / 256), dim3(256), 0, stream>>>(feat, mu, cat, N);
    mlp2(cat, mW1t + (size_t)l * 32768, mb1 + l * 128,
         mWst + (size_t)l * 16384, nullptr, mm, N, 256);
    // tensor interaction: Wt = swish(fij@tfW1+tfb1)*Cr ; s_agg = seg(dot(mu[src],vhat)*Wt)
    gemm(fij, tfW1t + (size_t)l * 4096, tfb1 + l * 128, Cr, eb1, E, 32, 1);
    zero(sg, (size_t)N * 128);
    k_scatter_tdot<<<dim3(E), dim3(128), 0, stream>>>(mu, vhat, eb1, src, dst, sg);
    gemm(sg, tWst + (size_t)l * 16384, tbs + l * 128, nullptr, vd, N, 128, 0);
    // combine + residual, then mu += dipole(l+1, v_total)
    k_combine<<<dim3((N * 128 + 255) / 256), dim3(256), 0, stream>>>(vt, mm, vd, feat, N);
    gemm(vt, Wdt + (size_t)(l + 1) * 16384, bd + (l + 1) * 128, nullptr, hn, N, 128, 1);
    k_scatter_dipole<<<dim3(E), dim3(128), 0, stream>>>(hn, Cr, vhat, src, dst, mu);
  }
}
